// MultiHeadAttention_7069516169746
// MI455X (gfx1250) — matrix operand order repaired and checked by reading, not compiled, not run on hardware
//
#include <hip/hip_runtime.h>
#include <hip/hip_bf16.h>
#include <math.h>

// ---------------------------------------------------------------------------
// Multi-head causal attention, MI455X (gfx1250), bf16 WMMA pipeline.
// B=4, S=2048, D=1024, H=16, Dh=64. Compute-bound (~100 GFLOP vs ~100MB I/O)
// => all matmuls via v_wmma_f32_16x16x32_bf16 (fp32 accumulate).
// Tiles sized to stay under the 256 directly-addressable VGPRs (64x64 spilled
// in round 2): GEMM uses a 64x32 register tile (8 WMMAs / 6KB loads), flash
// attention uses 16 query rows per wave (~130 VGPRs, no spills).
// ---------------------------------------------------------------------------

#define B_  4
#define S_  2048
#define D_  1024
#define H_  16
#define DH_ 64

typedef __bf16 v16bf __attribute__((ext_vector_type(16)));
typedef __bf16 v8bf  __attribute__((ext_vector_type(8)));
typedef float  v8f   __attribute__((ext_vector_type(8)));

// -------------------------- fragment loaders -------------------------------
// A-matrix 16x32 bf16 (ISA 7.12.2): lane = row (lane&15), half = lane>>4.
// Lane holds K = {half*8 .. +7} and {16+half*8 .. +7} (two 16B chunks).
__device__ __forceinline__ v16bf load_a_frag(const __bf16* base, int lda,
                                             int m0, int k0, int lane) {
  const int row  = m0 + (lane & 15);
  const int half = lane >> 4;
  const __bf16* p = base + (size_t)row * lda + k0 + half * 8;
  v8bf lo = *(const v8bf*)(p);
  v8bf hi = *(const v8bf*)(p + 16);
  v16bf r;
#pragma unroll
  for (int i = 0; i < 8; ++i) { r[i] = lo[i]; r[8 + i] = hi[i]; }
  return r;
}

// B-matrix 32x16 bf16 where B[k][n] = src[n][k] (src row-major [N,K]).
// Lane = col (lane&15), holds K = {half*8 .. +7} and {16+half*8 .. +7}.
__device__ __forceinline__ v16bf load_bT_frag(const __bf16* base, int ldb,
                                              int n0, int k0, int lane) {
  const int col = n0 + (lane & 15);
  const int kb  = k0 + (lane >> 4) * 8;
  const __bf16* p = base + (size_t)col * ldb + kb;
  v8bf lo = *(const v8bf*)(p);
  v8bf hi = *(const v8bf*)(p + 16);
  v16bf r;
#pragma unroll
  for (int i = 0; i < 8; ++i) { r[i] = lo[i]; r[8 + i] = hi[i]; }
  return r;
}

__device__ __forceinline__ v8f wmma_bf16(v16bf a, v16bf b, v8f c) {
  return __builtin_amdgcn_wmma_f32_16x16x32_bf16(
      false, a, false, b, (short)0, c, false, false);
}

// row reduction across the 16 lanes sharing a C-matrix row
__device__ __forceinline__ float rmax16(float v) {
  v = fmaxf(v, __shfl_xor(v, 1, 32));
  v = fmaxf(v, __shfl_xor(v, 2, 32));
  v = fmaxf(v, __shfl_xor(v, 4, 32));
  v = fmaxf(v, __shfl_xor(v, 8, 32));
  return v;
}
__device__ __forceinline__ float rsum16(float v) {
  v += __shfl_xor(v, 1, 32);
  v += __shfl_xor(v, 2, 32);
  v += __shfl_xor(v, 4, 32);
  v += __shfl_xor(v, 8, 32);
  return v;
}

// -------------------------- fp32 -> bf16 convert ---------------------------
__global__ void cvt_f32_bf16_kernel(const float* __restrict__ in,
                                    __bf16* __restrict__ out, int n) {
  int i = blockIdx.x * blockDim.x + threadIdx.x;
  if (i < n) out[i] = (__bf16)in[i];
}

// -------------------------- GEMM: y = x @ W^T + b --------------------------
// A [Mtot,Kdim] bf16 row-major, W [Ntot,Kdim] bf16 row-major, bias fp32.
// One wave per 64x32 output tile: 4x2 WMMA accumulators, K-step 32,
// 8 WMMAs per 6KB of fragment loads, ~180 VGPRs with pipelining (no spill).
// mode 0: bf16 out, split-head [B,H,S,Dh]            (Q, K)
// mode 1: bf16 out, split-head transposed [B,H,Dh,S] (V^T)
// mode 2: fp32 out, row-major [Mtot,Ntot]            (final projection)
__global__ void gemm_xwt_kernel(const __bf16* __restrict__ A,
                                const __bf16* __restrict__ W,
                                const float*  __restrict__ bias,
                                void* __restrict__ out,
                                int Mtot, int Kdim, int Ntot, int mode) {
  const int lane = threadIdx.x;
  const int n0 = blockIdx.x * 32;
  const int m0 = blockIdx.y * 64;

  v8f zero = {};
  v8f acc[4][2];
#pragma unroll
  for (int mi = 0; mi < 4; ++mi)
#pragma unroll
    for (int nj = 0; nj < 2; ++nj) acc[mi][nj] = zero;

  for (int k0 = 0; k0 < Kdim; k0 += 32) {
    v16bf a[4], bb[2];
#pragma unroll
    for (int mi = 0; mi < 4; ++mi)
      a[mi] = load_a_frag(A, Kdim, m0 + mi * 16, k0, lane);
#pragma unroll
    for (int nj = 0; nj < 2; ++nj)
      bb[nj] = load_bT_frag(W, Kdim, n0 + nj * 16, k0, lane);
#pragma unroll
    for (int mi = 0; mi < 4; ++mi)
#pragma unroll
      for (int nj = 0; nj < 2; ++nj)
        acc[mi][nj] = wmma_bf16(a[mi], bb[nj], acc[mi][nj]);
  }

  const int colL  = lane & 15;
  const int rbase = (lane >> 4) * 8;

#pragma unroll
  for (int mi = 0; mi < 4; ++mi) {
#pragma unroll
    for (int nj = 0; nj < 2; ++nj) {
      const int ncol = n0 + nj * 16 + colL;
      const float bv = bias[ncol];
#pragma unroll
      for (int r = 0; r < 8; ++r) {
        const int mrow = m0 + mi * 16 + rbase + r;
        const float val = acc[mi][nj][r] + bv;
        if (mode == 2) {
          ((float*)out)[(size_t)mrow * Ntot + ncol] = val;
        } else {
          const int b = mrow / S_, s = mrow % S_;
          const int h = ncol / DH_, d = ncol % DH_;
          size_t idx;
          if (mode == 0)
            idx = (((size_t)b * H_ + h) * S_ + s) * DH_ + d;   // [B,H,S,Dh]
          else
            idx = (((size_t)b * H_ + h) * DH_ + d) * S_ + s;   // [B,H,Dh,S]
          ((__bf16*)out)[idx] = (__bf16)val;
        }
      }
    }
  }
}

// -------------------------- flash attention --------------------------------
// grid: (S/16 query tiles, H, B); one wave (32 threads) per block.
// Q,K: [B,H,S,Dh] bf16.  Vt: [B,H,Dh,S] bf16.  Out: [B,S,D] bf16.
__global__ void flash_attn_kernel(const __bf16* __restrict__ Q,
                                  const __bf16* __restrict__ K,
                                  const __bf16* __restrict__ Vt,
                                  __bf16* __restrict__ Out) {
  __shared__ __bf16 lds_p[16 * 32];   // P tile staging for C->A transpose

  const int lane = threadIdx.x;
  const int qt = blockIdx.x, h = blockIdx.y, b = blockIdx.z;
  const size_t bh = (size_t)b * H_ + h;
  const __bf16* Qp = Q  + bh * (size_t)S_ * DH_;
  const __bf16* Kp = K  + bh * (size_t)S_ * DH_;
  const __bf16* Vp = Vt + bh * (size_t)DH_ * S_;
  const int m0 = qt * 16;

  const int colL  = lane & 15;
  const int rbase = (lane >> 4) * 8;
  const int half  = lane >> 4;
  const float scale = 0.125f;   // 1/sqrt(64)

  // Q fragments for K-dim 64 (two k32 steps), reused across all key blocks
  const v16bf qa0 = load_a_frag(Qp, DH_, m0, 0,  lane);
  const v16bf qa1 = load_a_frag(Qp, DH_, m0, 32, lane);

  v8f zero = {};
  v8f o[4]; o[0] = zero; o[1] = zero; o[2] = zero; o[3] = zero;
  float m_i[8], l_i[8];
#pragma unroll
  for (int r = 0; r < 8; ++r) { m_i[r] = -3.0e38f; l_i[r] = 0.0f; }

  const int nblocks = (m0 + 16 + 31) >> 5;   // causal: keys up to row m0+15
  for (int blk = 0; blk < nblocks; ++blk) {
    const int t0 = blk * 32;

    // S = Q K^T  (two 16-col tiles, K-dim 64)
    v16bf kb00 = load_bT_frag(Kp, DH_, t0,      0,  lane);
    v16bf kb01 = load_bT_frag(Kp, DH_, t0,      32, lane);
    v16bf kb10 = load_bT_frag(Kp, DH_, t0 + 16, 0,  lane);
    v16bf kb11 = load_bT_frag(Kp, DH_, t0 + 16, 32, lane);
    v8f s0 = zero, s1 = zero;
    s0 = wmma_bf16(qa0, kb00, s0);
    s0 = wmma_bf16(qa1, kb01, s0);
    s1 = wmma_bf16(qa0, kb10, s1);
    s1 = wmma_bf16(qa1, kb11, s1);

    // scale + causal mask + row max over this block
    float bmax[8];
#pragma unroll
    for (int r = 0; r < 8; ++r) {
      const int m = m0 + rbase + r;
      float v0 = s0[r] * scale;
      float v1 = s1[r] * scale;
      if (t0 + colL      > m) v0 = -1.0e30f;
      if (t0 + 16 + colL > m) v1 = -1.0e30f;
      s0[r] = v0; s1[r] = v1;
      bmax[r] = rmax16(fmaxf(v0, v1));
    }

    // online softmax update
    float alpha[8];
#pragma unroll
    for (int r = 0; r < 8; ++r) {
      const float mn = fmaxf(m_i[r], bmax[r]);
      alpha[r] = __expf(m_i[r] - mn);
      m_i[r] = mn;
      const float p0 = __expf(s0[r] - mn);
      const float p1 = __expf(s1[r] - mn);
      s0[r] = p0; s1[r] = p1;
      const float bsum = rsum16(p0 + p1);
      l_i[r] = l_i[r] * alpha[r] + bsum;
    }
#pragma unroll
    for (int j = 0; j < 4; ++j)
#pragma unroll
      for (int r = 0; r < 8; ++r) o[j][r] *= alpha[r];

    // transpose P (C-layout, fp32) -> A-layout bf16 via LDS
    __syncthreads();
#pragma unroll
    for (int r = 0; r < 8; ++r) {
      lds_p[(rbase + r) * 32 + colL]      = (__bf16)s0[r];
      lds_p[(rbase + r) * 32 + 16 + colL] = (__bf16)s1[r];
    }
    __syncthreads();
    {
      const int row = lane & 15;
      v8bf lo = *(const v8bf*)&lds_p[row * 32 + half * 8];
      v8bf hi = *(const v8bf*)&lds_p[row * 32 + 16 + half * 8];
      v16bf pa;
#pragma unroll
      for (int i = 0; i < 8; ++i) { pa[i] = lo[i]; pa[8 + i] = hi[i]; }

      // O += P V   (V^T layout [Dh,S] makes B-frags contiguous)
#pragma unroll
      for (int j = 0; j < 4; ++j) {
        v16bf vb = load_bT_frag(Vp, S_, j * 16, t0, lane);
        o[j] = wmma_bf16(pa, vb, o[j]);
      }
    }
  }

  // epilogue: normalize and store [B,S,D] bf16
  float inv[8];
#pragma unroll
  for (int r = 0; r < 8; ++r) inv[r] = 1.0f / l_i[r];
#pragma unroll
  for (int j = 0; j < 4; ++j) {
#pragma unroll
    for (int r = 0; r < 8; ++r) {
      const int s = m0 + rbase + r;
      const int col = h * DH_ + j * 16 + colL;
      Out[((size_t)b * S_ + s) * D_ + col] = (__bf16)(o[j][r] * inv[r]);
    }
  }
}

// ---------------------------------------------------------------------------
extern "C" void kernel_launch(void* const* d_in, const int* in_sizes, int n_in,
                              void* d_out, int out_size, void* d_ws, size_t ws_size,
                              hipStream_t stream) {
  (void)in_sizes; (void)n_in; (void)out_size; (void)ws_size;

  const float* x  = (const float*)d_in[0];
  const float* Wq = (const float*)d_in[1];
  const float* bq = (const float*)d_in[2];
  const float* Wk = (const float*)d_in[3];
  const float* bk = (const float*)d_in[4];
  const float* Wv = (const float*)d_in[5];
  const float* bv = (const float*)d_in[6];
  const float* Wo = (const float*)d_in[7];
  const float* bo = (const float*)d_in[8];

  const size_t nX = (size_t)B_ * S_ * D_;   // 8,388,608
  const size_t nW = (size_t)D_ * D_;        // 1,048,576

  // workspace layout (bf16 buffers)
  char* ws = (char*)d_ws;
  __bf16* xb  = (__bf16*)(ws);                         // [B*S, D]
  __bf16* Wqb = (__bf16*)(ws + nX * 2);                // [D, D]
  __bf16* Wkb = Wqb + nW;
  __bf16* Wvb = Wkb + nW;
  __bf16* Wob = Wvb + nW;
  __bf16* Qb  = Wob + nW;                              // [B,H,S,Dh]
  __bf16* Kb  = Qb + nX;                               // [B,H,S,Dh]
  __bf16* Vtb = Kb + nX;                               // [B,H,Dh,S]
  __bf16* AOb = Vtb + nX;                              // [B,S,D]

  // 1) convert inputs to bf16
  cvt_f32_bf16_kernel<<<(int)((nX + 255) / 256), 256, 0, stream>>>(x, xb, (int)nX);
  cvt_f32_bf16_kernel<<<(int)((nW + 255) / 256), 256, 0, stream>>>(Wq, Wqb, (int)nW);
  cvt_f32_bf16_kernel<<<(int)((nW + 255) / 256), 256, 0, stream>>>(Wk, Wkb, (int)nW);
  cvt_f32_bf16_kernel<<<(int)((nW + 255) / 256), 256, 0, stream>>>(Wv, Wvb, (int)nW);
  cvt_f32_bf16_kernel<<<(int)((nW + 255) / 256), 256, 0, stream>>>(Wo, Wob, (int)nW);

  // 2) Q/K/V projections (WMMA GEMM, one wave per 64x32 tile)
  const int Mtot = B_ * S_;
  dim3 ggrid(D_ / 32, Mtot / 64);   // (32, 128)
  gemm_xwt_kernel<<<ggrid, 32, 0, stream>>>(xb, Wqb, bq, (void*)Qb,  Mtot, D_, D_, 0);
  gemm_xwt_kernel<<<ggrid, 32, 0, stream>>>(xb, Wkb, bk, (void*)Kb,  Mtot, D_, D_, 0);
  gemm_xwt_kernel<<<ggrid, 32, 0, stream>>>(xb, Wvb, bv, (void*)Vtb, Mtot, D_, D_, 1);

  // 3) causal flash attention (WMMA QK^T and PV), 16 query rows per wave
  dim3 agrid(S_ / 16, H_, B_);      // (128, 16, 4)
  flash_attn_kernel<<<agrid, 32, 0, stream>>>(Qb, Kb, Vtb, AOb);

  // 4) output projection -> fp32 d_out
  gemm_xwt_kernel<<<ggrid, 32, 0, stream>>>(AOb, Wob, bo, d_out, Mtot, D_, D_, 2);
}
